// HatchingBlock_57964878626851
// MI455X (gfx1250) — matrix operand order repaired and checked by reading, not compiled, not run on hardware
//
#include <hip/hip_runtime.h>
#include <hip/hip_bf16.h>
#include <math.h>

typedef __attribute__((ext_vector_type(16))) __bf16 v16bf;
typedef __attribute__((ext_vector_type(8)))  __bf16 v8bf;
typedef __attribute__((ext_vector_type(8)))  float  v8f;

namespace {
constexpr int B_  = 2;
constexpr int T_  = 512;
constexpr int D_  = 256;
constexpr int NH_ = 4;
constexpr int N_  = 8192;
constexpr int HN_ = NH_ * N_;          // 32768
constexpr int KSPLIT_ = 8;             // split-K factor for the MLP GEMM

constexpr float BETA_    = 0.09516258196404048f;  // 1 - exp(-1/10)
constexpr float A_PLUS_  = 0.01f;
constexpr float INV_TAU_ = 1.0f / 20.0f;
constexpr float LN_EPS_  = 1e-5f;
constexpr float TWO_PI_  = 6.283185307179586f;
constexpr float INV_2PI_ = 0.15915494309189535f;
}

// ---- WMMA fragment loaders (CDNA5 16-bit layouts, wave32) -----------------
// A 16x32: lane L -> row M=L&15; lanes 0-15 hold K {0..7,16..23}, lanes 16-31
// hold K {8..15,24..31}  => two 16-byte contiguous chunks per lane.
__device__ __forceinline__ v16bf load_frag_a(const __bf16* A, int lda,
                                             int row0, int k0, int lane) {
  const int m = lane & 15, half = lane >> 4;
  const __bf16* p = A + (size_t)(row0 + m) * lda + (k0 + half * 8);
  v8bf lo = *(const v8bf*)(p);
  v8bf hi = *(const v8bf*)(p + 16);
  v16bf r;
#pragma unroll
  for (int i = 0; i < 8; ++i) { r[i] = lo[i]; r[i + 8] = hi[i]; }
  return r;
}
// B 32x16: lane L -> col N=L&15; lanes 0-15 hold K {0..7,16..23}, lanes 16-31
// hold K {8..15,24..31}. Operand stored transposed (col-major => contiguous K per column).
__device__ __forceinline__ v16bf load_frag_b(const __bf16* Bt, int ldb,
                                             int col0, int k0, int lane) {
  const int n = lane & 15, half = lane >> 4;
  const __bf16* p = Bt + (size_t)(col0 + n) * ldb + (k0 + half * 8);
  v8bf lo = *(const v8bf*)(p);
  v8bf hi = *(const v8bf*)(p + 16);
  v16bf r;
#pragma unroll
  for (int i = 0; i < 8; ++i) { r[i] = lo[i]; r[i + 8] = hi[i]; }
  return r;
}

// ---- Prep: xs fp32 -> bf16 (row-major + transposed copies) ----------------
__global__ void k_prep_xs(const float* __restrict__ x,
                          __bf16* __restrict__ xs_bf,
                          __bf16* __restrict__ xsT_bf) {
  int idx = blockIdx.x * 256 + threadIdx.x;            // B*T*D
  int d = idx % D_;
  int t = (idx / D_) % T_;
  int b = idx / (T_ * D_);
  float v = x[idx];
  xs_bf[idx] = (__bf16)v;
  xsT_bf[((size_t)b * D_ + d) * T_ + t] = (__bf16)v;
}

// ---- LDS 32x32 tiled transpose: (H,R,C) fp32 -> (H,C,R) bf16 --------------
// Coalesced b32 reads and coalesced b16 writes; 33-wide rows avoid bank
// conflicts on the 64-bank LDS.
__global__ void k_transpose_bf(const float* __restrict__ in,
                               __bf16* __restrict__ out,
                               int R, int C, int rTiles) {
  __shared__ __bf16 tl[32][33];
  const int tx = threadIdx.x & 31;
  const int ty = threadIdx.x >> 5;                     // 0..7
  const int h  = blockIdx.y / rTiles;
  const int rt = blockIdx.y % rTiles;
  const int cb = blockIdx.x;
  const size_t ibase = (size_t)h * R * C;
  const size_t obase = (size_t)h * C * R;
#pragma unroll
  for (int i = 0; i < 4; ++i) {
    int r = rt * 32 + ty + i * 8;
    int c = cb * 32 + tx;
    tl[ty + i * 8][tx] = (__bf16)in[ibase + (size_t)r * C + c];
  }
  __syncthreads();
#pragma unroll
  for (int i = 0; i < 4; ++i) {
    int orow = cb * 32 + ty + i * 8;                   // original column
    int ocol = rt * 32 + tx;                           // original row
    out[obase + (size_t)orow * R + ocol] = tl[tx][ty + i * 8];
  }
}

// ---- 1) x_latent GEMM (1x4 N-blocked) + LIF neuron + RoPE epilogue --------
__global__ void k_encode(const __bf16* __restrict__ xs_bf,
                         const __bf16* __restrict__ encT,
                         float* __restrict__ membrane,
                         float* __restrict__ trace,
                         __bf16* __restrict__ QR,
                         __bf16* __restrict__ xsp) {
  const int lane = threadIdx.x & 31;
  const int wave = threadIdx.x >> 5;
  const unsigned tile = blockIdx.x * 8u + wave;        // 32768 tile-quads
  const int bh  = tile >> 12;
  const int rem = tile & 4095;
  const int tT  = rem >> 7;                            // 32
  const int tNq = rem & 127;                           // 128 quads of 64 cols
  const int b = bh >> 2, h = bh & 3;

  const __bf16* A  = xs_bf + (size_t)b * T_ * D_;
  const __bf16* Bt = encT + (size_t)h * N_ * D_;

  v8f acc[4] = {};
#pragma unroll
  for (int k0 = 0; k0 < D_; k0 += 32) {
    v16bf a = load_frag_a(A, D_, tT * 16, k0, lane);   // reused across 4 tiles
#pragma unroll
    for (int s = 0; s < 4; ++s) {
      v16bf bb = load_frag_b(Bt, D_, tNq * 64 + s * 16, k0, lane);
      acc[s] = __builtin_amdgcn_wmma_f32_16x16x32_bf16(false, a, false, bb,
                                                       (short)0, acc[s],
                                                       false, false);
    }
  }

  const int half = lane >> 4;
  const int colL = lane & 15;
#pragma unroll
  for (int s = 0; s < 4; ++s) {
    const int nIdx = tNq * 64 + s * 16 + colL;
    const int q = (nIdx >> 1) << 1;                    // theta = 2^16
    const float f = exp2f(-16.0f * (float)q / (float)N_) * INV_2PI_;
    const int odd = nIdx & 1;
#pragma unroll
    for (int r = 0; r < 8; ++r) {
      const int t = tT * 16 + r + half * 8;
      float lat   = acc[s][r];
      float mem   = BETA_ * lat;
      float spike = mem > 1.0f ? 1.0f : 0.0f;
      mem -= spike;                                    // V_TH - V_RESET = 1
      float xv = spike * fmaxf(lat, 0.0f);
      float ph = (float)t * f;
      ph = (ph - floorf(ph)) * TWO_PI_;
      float sn = __sinf(ph), cs = __cosf(ph);
      float partner = __shfl_xor(xv, 1, 32);           // pair (2k,2k+1)
      float vrot = odd ? partner : -partner;
      float qr = xv * cs + vrot * sn;
      size_t o = ((size_t)bh * T_ + t) * N_ + nIdx;
      membrane[o] = mem;
      trace[o]    = spike;
      QR[o]  = (__bf16)qr;
      xsp[o] = (__bf16)xv;
    }
  }
}

// ---- gate = row-sum of trace over N ---------------------------------------
__global__ void k_gate(const float* __restrict__ trace,
                       float* __restrict__ gateSum) {
  __shared__ float red[256];
  const int row = blockIdx.x;                          // B*NH*T
  const float* p = trace + (size_t)row * N_;
  float s = 0.f;
  for (int i = threadIdx.x; i < N_; i += 256) s += p[i];
  red[threadIdx.x] = s;
  __syncthreads();
  for (int off = 128; off > 0; off >>= 1) {
    if (threadIdx.x < off) red[threadIdx.x] += red[threadIdx.x + off];
    __syncthreads();
  }
  if (threadIdx.x == 0) gateSum[row] = red[0];
}

// ---- 2) scores = QR·QRt (1x4 S-blocked) + STDP / causal / gate ------------
__global__ void k_scores(const __bf16* __restrict__ QR,
                         const float* __restrict__ gateSum,
                         const float* __restrict__ stdp_scale,
                         __bf16* __restrict__ scores) {
  const int lane = threadIdx.x & 31;
  const int wave = threadIdx.x >> 5;
  const unsigned tile = blockIdx.x * 8u + wave;        // 2048 tile-quads
  const int bh  = tile >> 8;
  const int rem = tile & 255;
  const int tT  = rem >> 3;                            // 32
  const int tSq = rem & 7;                             // 8 quads of 64 cols

  const int half = lane >> 4;
  const int colL = lane & 15;

  if (tSq * 4 > tT) {                                  // whole quad above diag
#pragma unroll
    for (int s = 0; s < 4; ++s) {
      const int sIdx = tSq * 64 + s * 16 + colL;
#pragma unroll
      for (int r = 0; r < 8; ++r) {
        const int t = tT * 16 + r + half * 8;
        scores[((size_t)bh * T_ + t) * T_ + sIdx] = (__bf16)0.0f;
      }
    }
    return;
  }

  const __bf16* base = QR + (size_t)bh * T_ * N_;
  v8f acc[4] = {};
  for (int k0 = 0; k0 < N_; k0 += 32) {                // K = 8192
    __builtin_prefetch((const void*)(base + (size_t)(tT * 16 + colL) * N_
                                     + k0 + 256), 0, 1);
    v16bf a = load_frag_a(base, N_, tT * 16, k0, lane);
#pragma unroll
    for (int s = 0; s < 4; ++s) {
      v16bf bb = load_frag_b(base, N_, tSq * 64 + s * 16, k0, lane);
      acc[s] = __builtin_amdgcn_wmma_f32_16x16x32_bf16(false, a, false, bb,
                                                       (short)0, acc[s],
                                                       false, false);
    }
  }

  const float sv = stdp_scale[0];
#pragma unroll
  for (int s = 0; s < 4; ++s) {
    const int sIdx = tSq * 64 + s * 16 + colL;
#pragma unroll
    for (int r = 0; r < 8; ++r) {
      const int t = tT * 16 + r + half * 8;
      float val = 0.0f;
      if (sIdx < t) {                                  // tril(-1); dt>0 => LTP
        float kern = A_PLUS_ * expf(-(float)(t - sIdx) * INV_TAU_) * sv;
        float g = gateSum[(size_t)bh * T_ + t] * (1.0f / (float)N_);
        float sig = 1.0f / (1.0f + expf(-g));
        val = acc[s][r] * (1.0f + kern) * (0.5f + 0.5f * sig);
      }
      scores[((size_t)bh * T_ + t) * T_ + sIdx] = (__bf16)val;
    }
  }
}

// ---- 3) yKV = scores · xs (1x4 D-blocked) ---------------------------------
__global__ void k_ykv(const __bf16* __restrict__ scores,
                      const __bf16* __restrict__ xsT,
                      float* __restrict__ yKV) {
  const int lane = threadIdx.x & 31;
  const int wave = threadIdx.x >> 5;
  const unsigned tile = blockIdx.x * 8u + wave;        // 1024 tile-quads
  const int bh  = tile >> 7;
  const int rem = tile & 127;
  const int tT  = rem >> 2;                            // 32
  const int tDq = rem & 3;                             // 4 quads of 64 cols
  const int b = bh >> 2;

  const __bf16* A  = scores + (size_t)bh * T_ * T_;
  const __bf16* Bt = xsT + (size_t)b * D_ * T_;

  v8f acc[4] = {};
#pragma unroll
  for (int k0 = 0; k0 < T_; k0 += 32) {
    v16bf a = load_frag_a(A, T_, tT * 16, k0, lane);
#pragma unroll
    for (int s = 0; s < 4; ++s) {
      v16bf bb = load_frag_b(Bt, T_, tDq * 64 + s * 16, k0, lane);
      acc[s] = __builtin_amdgcn_wmma_f32_16x16x32_bf16(false, a, false, bb,
                                                       (short)0, acc[s],
                                                       false, false);
    }
  }
  const int half = lane >> 4;
  const int colL = lane & 15;
#pragma unroll
  for (int s = 0; s < 4; ++s) {
    const int dIdx = tDq * 64 + s * 16 + colL;
#pragma unroll
    for (int r = 0; r < 8; ++r) {
      const int t = tT * 16 + r + half * 8;
      yKV[((size_t)bh * T_ + t) * D_ + dIdx] = acc[s][r];
    }
  }
}

// ---- LN over D for yKV ----------------------------------------------------
__global__ void k_ln_ykv(const float* __restrict__ yKV,
                         __bf16* __restrict__ yKV_bf) {
  __shared__ float sm[256], sq[256];
  const int row = blockIdx.x;                          // B*NH*T
  float v = yKV[(size_t)row * D_ + threadIdx.x];
  sm[threadIdx.x] = v;
  sq[threadIdx.x] = v * v;
  __syncthreads();
  for (int off = 128; off > 0; off >>= 1) {
    if (threadIdx.x < off) {
      sm[threadIdx.x] += sm[threadIdx.x + off];
      sq[threadIdx.x] += sq[threadIdx.x + off];
    }
    __syncthreads();
  }
  float mu  = sm[0] / D_;
  float var = sq[0] / D_ - mu * mu;
  yKV_bf[(size_t)row * D_ + threadIdx.x] =
      (__bf16)((v - mu) * rsqrtf(var + LN_EPS_));
}

// ---- 4) y_latent GEMM (1x4) fused with x_sparse*relu -> xy layout ---------
__global__ void k_ylat(const __bf16* __restrict__ yKV_bf,
                       const __bf16* __restrict__ encVT,
                       const __bf16* __restrict__ xsp,
                       __bf16* __restrict__ xy) {
  const int lane = threadIdx.x & 31;
  const int wave = threadIdx.x >> 5;
  const unsigned tile = blockIdx.x * 8u + wave;        // 32768 tile-quads
  const int bh  = tile >> 12;
  const int rem = tile & 4095;
  const int tT  = rem >> 7;
  const int tNq = rem & 127;
  const int b = bh >> 2, h = bh & 3;

  const __bf16* A  = yKV_bf + (size_t)bh * T_ * D_;
  const __bf16* Bt = encVT + (size_t)h * N_ * D_;

  v8f acc[4] = {};
#pragma unroll
  for (int k0 = 0; k0 < D_; k0 += 32) {
    v16bf a = load_frag_a(A, D_, tT * 16, k0, lane);
#pragma unroll
    for (int s = 0; s < 4; ++s) {
      v16bf bb = load_frag_b(Bt, D_, tNq * 64 + s * 16, k0, lane);
      acc[s] = __builtin_amdgcn_wmma_f32_16x16x32_bf16(false, a, false, bb,
                                                       (short)0, acc[s],
                                                       false, false);
    }
  }
  const int half = lane >> 4;
  const int colL = lane & 15;
#pragma unroll
  for (int s = 0; s < 4; ++s) {
    const int nIdx = tNq * 64 + s * 16 + colL;
#pragma unroll
    for (int r = 0; r < 8; ++r) {
      const int t = tT * 16 + r + half * 8;
      float xv = (float)xsp[((size_t)bh * T_ + t) * N_ + nIdx];
      float o  = xv * fmaxf(acc[s][r], 0.0f);
      // transposed write to (b, t, h*N + n): MLP A-operand row-major over K
      xy[((size_t)b * T_ + t) * HN_ + (size_t)h * N_ + nIdx] = (__bf16)o;
    }
  }
}

// ---- 5) yMLP = xy · decoder (1x4 D-blocked, split-K x8) -------------------
__global__ void k_mlp(const __bf16* __restrict__ xy,
                      const __bf16* __restrict__ decT,
                      float* __restrict__ yMLPp) {   // [KSPLIT][B*T][D]
  const int lane = threadIdx.x & 31;
  const int wave = threadIdx.x >> 5;
  const unsigned tile = blockIdx.x * 8u + wave;        // 2048 work items
  const int tM  = tile >> 5;                           // (B*T)/16 = 64
  const int rem = tile & 31;
  const int tDq = rem >> 3;                            // 4 quads of 64 cols
  const int ks  = rem & 7;                             // K-slice

  const int kBeg = ks * (HN_ / KSPLIT_);
  const int kEnd = kBeg + (HN_ / KSPLIT_);

  v8f acc[4] = {};
  for (int k0 = kBeg; k0 < kEnd; k0 += 32) {           // 4096 per slice
    v16bf a = load_frag_a(xy, HN_, tM * 16, k0, lane);
#pragma unroll
    for (int s = 0; s < 4; ++s) {
      v16bf bb = load_frag_b(decT, HN_, tDq * 64 + s * 16, k0, lane);
      acc[s] = __builtin_amdgcn_wmma_f32_16x16x32_bf16(false, a, false, bb,
                                                       (short)0, acc[s],
                                                       false, false);
    }
  }
  const int half = lane >> 4;
  const int colL = lane & 15;
  float* outp = yMLPp + (size_t)ks * B_ * T_ * D_;
#pragma unroll
  for (int s = 0; s < 4; ++s) {
    const int dIdx = tDq * 64 + s * 16 + colL;
#pragma unroll
    for (int r = 0; r < 8; ++r) {
      const int row = tM * 16 + r + half * 8;
      outp[(size_t)row * D_ + dIdx] = acc[s][r];
    }
  }
}

// ---- 6) final: sum split-K partials; y = LN(yMLP); out = LN(x + y) --------
__global__ void k_final(const float* __restrict__ yMLPp,
                        const float* __restrict__ x,
                        float* __restrict__ out) {
  __shared__ float s1[256], s2[256];
  const int row = blockIdx.x;                          // B*T
  const int tid = threadIdx.x;
  float ym = 0.0f;
#pragma unroll
  for (int ks = 0; ks < KSPLIT_; ++ks)
    ym += yMLPp[(size_t)ks * B_ * T_ * D_ + (size_t)row * D_ + tid];
  s1[tid] = ym; s2[tid] = ym * ym;
  __syncthreads();
  for (int off = 128; off > 0; off >>= 1) {
    if (tid < off) { s1[tid] += s1[tid + off]; s2[tid] += s2[tid + off]; }
    __syncthreads();
  }
  float mu  = s1[0] / D_;
  float var = s2[0] / D_ - mu * mu;
  float y   = (ym - mu) * rsqrtf(var + LN_EPS_);
  __syncthreads();
  float z = x[(size_t)row * D_ + tid] + y;
  s1[tid] = z; s2[tid] = z * z;
  __syncthreads();
  for (int off = 128; off > 0; off >>= 1) {
    if (tid < off) { s1[tid] += s1[tid + off]; s2[tid] += s2[tid + off]; }
    __syncthreads();
  }
  mu  = s1[0] / D_;
  var = s2[0] / D_ - mu * mu;
  out[(size_t)row * D_ + tid] = (z - mu) * rsqrtf(var + LN_EPS_);
}

// ---------------------------------------------------------------------------
extern "C" void kernel_launch(void* const* d_in, const int* in_sizes, int n_in,
                              void* d_out, int out_size, void* d_ws, size_t ws_size,
                              hipStream_t stream) {
  (void)in_sizes; (void)n_in; (void)out_size; (void)ws_size;
  const float* x    = (const float*)d_in[0];
  const float* enc  = (const float*)d_in[1];
  const float* encv = (const float*)d_in[2];
  const float* dec  = (const float*)d_in[3];
  const float* stdp = (const float*)d_in[4];

  float* out      = (float*)d_out;                         // (B,1,T,D)
  float* membrane = out + (size_t)B_ * T_ * D_;            // (B,NH,T,N)
  float* trace    = membrane + (size_t)B_ * NH_ * T_ * N_; // (B,NH,T,N)

  char* w = (char*)d_ws;
  size_t o = 0;
  auto take = [&](size_t bytes) -> char* {
    char* p = w + o;
    o += (bytes + 255) & ~(size_t)255;
    return p;
  };
  __bf16* xs_bf   = (__bf16*)take((size_t)B_ * T_ * D_ * 2);
  __bf16* xsT_bf  = (__bf16*)take((size_t)B_ * D_ * T_ * 2);
  __bf16* encT    = (__bf16*)take((size_t)NH_ * N_ * D_ * 2);
  __bf16* encVT   = (__bf16*)take((size_t)NH_ * N_ * D_ * 2);
  __bf16* decT    = (__bf16*)take((size_t)D_ * HN_ * 2);
  __bf16* QR      = (__bf16*)take((size_t)B_ * NH_ * T_ * N_ * 2);
  __bf16* xsp     = (__bf16*)take((size_t)B_ * NH_ * T_ * N_ * 2);
  __bf16* scores  = (__bf16*)take((size_t)B_ * NH_ * T_ * T_ * 2);
  float*  yKV     = (float*)take((size_t)B_ * NH_ * T_ * D_ * 4);
  __bf16* yKV_bf  = (__bf16*)take((size_t)B_ * NH_ * T_ * D_ * 2);
  __bf16* xy      = (__bf16*)take((size_t)B_ * T_ * HN_ * 2);
  float*  yMLPp   = (float*)take((size_t)KSPLIT_ * B_ * T_ * D_ * 4);
  float*  gateSum = (float*)take((size_t)B_ * NH_ * T_ * 4);

  // weight/value prep: bf16 cast + coalesced LDS-tile transposes
  k_prep_xs<<<(B_ * T_ * D_) / 256, 256, 0, stream>>>(x, xs_bf, xsT_bf);
  k_transpose_bf<<<dim3(N_ / 32, (D_ / 32) * NH_), 256, 0, stream>>>(
      enc, encT, D_, N_, D_ / 32);
  k_transpose_bf<<<dim3(N_ / 32, (D_ / 32) * NH_), 256, 0, stream>>>(
      encv, encVT, D_, N_, D_ / 32);
  k_transpose_bf<<<dim3(D_ / 32, HN_ / 32), 256, 0, stream>>>(
      dec, decT, HN_, D_, HN_ / 32);

  k_encode<<<4096, 256, 0, stream>>>(xs_bf, encT, membrane, trace, QR, xsp);
  k_gate  <<<B_ * NH_ * T_, 256, 0, stream>>>(trace, gateSum);
  k_scores<<<256, 256, 0, stream>>>(QR, gateSum, stdp, scores);
  k_ykv   <<<128, 256, 0, stream>>>(scores, xsT_bf, yKV);
  k_ln_ykv<<<B_ * NH_ * T_, 256, 0, stream>>>(yKV, yKV_bf);
  k_ylat  <<<4096, 256, 0, stream>>>(yKV_bf, encVT, xsp, xy);
  k_mlp   <<<256, 256, 0, stream>>>(xy, decT, yMLPp);
  k_final <<<B_ * T_, 256, 0, stream>>>(yMLPp, x, out);
}
